// GAT_27951647162777
// MI455X (gfx1250) — matrix operand order repaired and checked by reading, not compiled, not run on hardware
//
#include <hip/hip_runtime.h>
#include <hip/hip_bf16.h>

typedef __attribute__((ext_vector_type(16))) __bf16 v16bf;
typedef __attribute__((ext_vector_type(8)))  float  v8f;

union AB16 { unsigned short u[16]; uint4 q[2]; v16bf v; };

__device__ __forceinline__ unsigned short f2bf(float f) {
    unsigned u = __float_as_uint(f);
    unsigned r = u + 0x7FFFu + ((u >> 16) & 1u);
    return (unsigned short)(r >> 16);
}
__device__ __forceinline__ float bf2f(unsigned short h) {
    return __uint_as_float(((unsigned)h) << 16);
}

// ---------------- packing kernels ----------------
__global__ void cvt_f32_bf16(const float* __restrict__ in, unsigned short* __restrict__ out, long n) {
    long i = (long)blockIdx.x * blockDim.x + threadIdx.x;
    if (i < n) out[i] = f2bf(in[i]);
}

// W1 [H][F][O] -> W1T [H][O][F]
__global__ void pack_W1T(const float* __restrict__ W1, unsigned short* __restrict__ W1T,
                         int H, int F, int O) {
    long i = (long)blockIdx.x * blockDim.x + threadIdx.x;
    long tot = (long)H * F * O;
    if (i >= tot) return;
    int f = (int)(i % F);
    int o = (int)((i / F) % O);
    int h = (int)(i / ((long)F * O));
    W1T[i] = f2bf(W1[((long)h * F + f) * O + o]);
}

// W2 [F][C] -> W2T [48][F], zero-padded cols >= C
__global__ void pack_W2T(const float* __restrict__ W2, unsigned short* __restrict__ W2T,
                         int F, int C, int Cpad) {
    long i = (long)blockIdx.x * blockDim.x + threadIdx.x;
    if (i >= (long)Cpad * F) return;
    int f = (int)(i % F);
    int c = (int)(i / F);
    W2T[i] = (c < C) ? f2bf(W2[(long)f * C + c]) : (unsigned short)0;
}

// ---------------- WMMA GEMM:  outT[N][M] = (A[M,K] @ B[K,N])^T, bf16 in/out
// grid (M/16, Ntot/16), block 32 (one wave per 16x16-output column group).
// 2-stage software pipeline: next K-fragments load while current WMMA executes.
__global__ void gemm_bf16_wmma(const unsigned short* __restrict__ A,
                               const unsigned short* __restrict__ BT,   // [Ntot][K]
                               unsigned short* __restrict__ outT,       // [Ntot][pitchOut]
                               int K, int pitchOut) {
    int lane = threadIdx.x;
    int nt   = blockIdx.y;
    int row0 = blockIdx.x * 16;
    int l15  = lane & 15;
    int hi8  = (lane >> 4) << 3;
    int hi16 = hi8 << 1;
    const unsigned short* Ar = A  + (size_t)(row0 + l15) * K + hi8;
    const unsigned short* Br = BT + (size_t)(nt * 16 + l15) * K + hi8;

    v8f c = {};
    AB16 af, bf;
    af.q[0] = *(const uint4*)(Ar);      af.q[1] = *(const uint4*)(Ar + 16);
    bf.q[0] = *(const uint4*)(Br);      bf.q[1] = *(const uint4*)(Br + 16);
    for (int kb = 0; kb < K; kb += 32) {
        AB16 afn = af, bfn = bf;
        int kn = kb + 32;
        if (kn < K) {                    // prefetch next fragments before consuming current
            afn.q[0] = *(const uint4*)(Ar + kn);      afn.q[1] = *(const uint4*)(Ar + kn + 16);
            bfn.q[0] = *(const uint4*)(Br + kn);      bfn.q[1] = *(const uint4*)(Br + kn + 16);
        }
        c = __builtin_amdgcn_wmma_f32_16x16x32_bf16(false, af.v, false, bf.v,
                                                    (short)0, c, false, false);
        af = afn; bf = bfn;
    }
    union { unsigned short u[8]; uint4 q; } o;
#pragma unroll
    for (int r = 0; r < 8; ++r) o.u[r] = f2bf(c[r]);
    *(uint4*)(outT + (size_t)(nt * 16 + l15) * pitchOut + row0 + hi8) = o.q;
}

// ---------------- s_src / s_tgt dots from WhT [rowsW][pitchW] ----------------
__global__ void sdot_k(const unsigned short* __restrict__ WhT,
                       const float* __restrict__ a_src, const float* __restrict__ a_tgt,
                       float* __restrict__ s_src, float* __restrict__ s_tgt,
                       int Fdot, int Nn, int pitchW, int rowsW, int aStride) {
    int head = blockIdx.y;
    int i = blockIdx.x * blockDim.x + threadIdx.x;
    if (i >= Nn) return;
    const unsigned short* Wh = WhT + (size_t)head * rowsW * pitchW;
    const float* as = a_src + (size_t)head * aStride;
    const float* at = a_tgt + (size_t)head * aStride;
    float accs = 0.f, acct = 0.f;
    for (int f = 0; f < Fdot; ++f) {
        float w = bf2f(Wh[(size_t)f * pitchW + i]);
        accs += as[f] * w;
        acct += at[f] * w;
    }
    s_src[(size_t)head * Nn + i] = accs;
    s_tgt[(size_t)head * Nn + i] = acct;
}

// ---------------- per-row neighbor max of s_tgt (all heads, adj read once) ----------------
template<int HN>
__global__ void rowmax_k(const int* __restrict__ adj, const float* __restrict__ s_tgt,
                         float* __restrict__ mmax, int Nn) {
    int i = blockIdx.x;
    int tid = threadIdx.x;
    float mx[HN];
#pragma unroll
    for (int h = 0; h < HN; ++h) mx[h] = -3.0e38f;
    const int* ar = adj + (size_t)i * Nn;
    for (int j = tid; j < Nn; j += blockDim.x) {
        if (ar[j] > 0) {
#pragma unroll
            for (int h = 0; h < HN; ++h) mx[h] = fmaxf(mx[h], s_tgt[(size_t)h * Nn + j]);
        }
    }
    __shared__ float red[256];
#pragma unroll
    for (int h = 0; h < HN; ++h) {
        red[tid] = mx[h];
        __syncthreads();
        for (int s = 128; s > 0; s >>= 1) {
            if (tid < s) red[tid] = fmaxf(red[tid], red[tid + s]);
            __syncthreads();
        }
        if (tid == 0) mmax[(size_t)h * Nn + i] = red[0];
        __syncthreads();
    }
}

// ---------------- fused GAT attention: softmax(p) @ Wh via WMMA, never materialized.
// grid (Nn/16, heads), block 32*WV. WV waves split the J (neighbor) loop; the softmax
// max is precomputed (leaky-relu monotone), so partial (c, l) just add -> LDS combine.
// MODE 0: bf16 ELU -> h1b [Nn][outPitch], cols head*NT*16 + n. MODE 1: fp32 out, n<outPitch.
template<int NT, int MODE, int WV>
__global__ __launch_bounds__(32 * WV)
void gat_attn_wmma(const int* __restrict__ adj,
                   const float* __restrict__ s_src,
                   const float* __restrict__ s_tgt,
                   const float* __restrict__ mmax,
                   const unsigned short* __restrict__ WhT, // [H][NT*16][pitchW]
                   void* __restrict__ outp,
                   int Nn, int pitchW, int outPitch) {
    int tid  = threadIdx.x;
    int lane = tid & 31;
    int wave = tid >> 5;
    int head = blockIdx.y;
    int row0 = blockIdx.x * 16;
    int l15  = lane & 15;
    int hi8  = (lane >> 4) << 3;
    int hi16 = hi8 << 1;

    const float* ss = s_src + (size_t)head * Nn;
    const float* st = s_tgt + (size_t)head * Nn;
    const float* mm = mmax  + (size_t)head * Nn;
    const unsigned short* Wh = WhT + (size_t)head * (NT * 16) * pitchW;

    int arow = row0 + l15;                 // A-fragment row for this lane
    float si = ss[arow];
    float mi;
    { float tmx = si + mm[arow]; mi = tmx > 0.f ? tmx : 0.2f * tmx; }  // true row max

    v8f c[NT] = {};
    float lsum = 0.f;
    const int* adjr = adj + (size_t)arow * Nn;
    const int step = 32 * WV;

    for (int Jb = wave * 32; Jb < Nn; Jb += step) {
        bool tail = (Jb + 32 > Nn);        // last 16-col sliver (6000 = 187*32 + 16)
        int base = Jb + hi8;
        // speculative prefetch of this wave's next chunk (OOB prefetch is dropped by HW)
        __builtin_prefetch(adjr + base + step, 0, 0);
        __builtin_prefetch(st + base + step, 0, 0);

        int av[16]; float sv[16];
        *(int4*)&av[0]   = *(const int4*)(adjr + base);
        *(int4*)&av[4]   = *(const int4*)(adjr + base + 4);
        *(float4*)&sv[0] = *(const float4*)(st + base);
        *(float4*)&sv[4] = *(const float4*)(st + base + 4);
        if (!tail) {
            *(int4*)&av[8]    = *(const int4*)(adjr + base + 16);
            *(int4*)&av[12]   = *(const int4*)(adjr + base + 20);
            *(float4*)&sv[8]  = *(const float4*)(st + base + 16);
            *(float4*)&sv[12] = *(const float4*)(st + base + 20);
        } else {
#pragma unroll
            for (int e = 8; e < 16; ++e) { av[e] = 0; sv[e] = 0.f; }
        }
        // batch all B tiles first: one clause + one wait, then back-to-back WMMAs
        AB16 bfr[NT];
#pragma unroll
        for (int t = 0; t < NT; ++t) {
            const unsigned short* bp = Wh + (size_t)(t * 16 + l15) * pitchW + Jb + hi8;
            bfr[t].q[0] = *(const uint4*)(bp);
            bfr[t].q[1] = *(const uint4*)(bp + 16);  // tail lands in finite pad cols
        }
        AB16 af;
        float lacc = 0.f;
#pragma unroll
        for (int e = 0; e < 16; ++e) {
            float p = 0.f;
            if (av[e] > 0) {
                float ev = si + sv[e];
                ev = ev > 0.f ? ev : 0.2f * ev;   // leaky relu
                p = __expf(ev - mi);              // <= 1; masked entries exactly 0
            }
            lacc += p;
            af.u[e] = f2bf(p);
        }
        lsum += lacc;
#pragma unroll
        for (int t = 0; t < NT; ++t)
            c[t] = __builtin_amdgcn_wmma_f32_16x16x32_bf16(false, af.v, false, bfr[t].v,
                                                           (short)0, c[t], false, false);
    }

    // per-wave row totals (both lane halves of each row)
    float ltot = lsum + __shfl_xor(lsum, 16, 32);

    // combine the WV waves' partial accumulators through LDS
    __shared__ float redc[WV][NT][8][32];
    __shared__ float redl[WV][32];
#pragma unroll
    for (int t = 0; t < NT; ++t)
#pragma unroll
        for (int r = 0; r < 8; ++r) redc[wave][t][r][lane] = c[t][r];
    redl[wave][lane] = ltot;
    __syncthreads();

    if (wave == 0) {
        float lt = 0.f;
#pragma unroll
        for (int w = 0; w < WV; ++w) lt += redl[w][lane];
        float denom[8];
#pragma unroll
        for (int r = 0; r < 8; ++r) denom[r] = __shfl(lt, hi8 + r, 32);  // C-tile row M=r+hi8

        if (MODE == 0) {
            unsigned short* out = (unsigned short*)outp;
            int colBase = head * (NT * 16);
#pragma unroll
            for (int t = 0; t < NT; ++t) {
#pragma unroll
                for (int r = 0; r < 8; ++r) {
                    float v = 0.f;
#pragma unroll
                    for (int w = 0; w < WV; ++w) v += redc[w][t][r][lane];
                    v /= denom[r];
                    v = v > 0.f ? v : (__expf(v) - 1.f);          // ELU
                    out[(size_t)(row0 + hi8 + r) * outPitch + colBase + t * 16 + l15] = f2bf(v);
                }
            }
        } else {
            float* out = (float*)outp;
#pragma unroll
            for (int t = 0; t < NT; ++t) {
                int n = t * 16 + l15;
                if (n < outPitch) {
#pragma unroll
                    for (int r = 0; r < 8; ++r) {
                        float v = 0.f;
#pragma unroll
                        for (int w = 0; w < WV; ++w) v += redc[w][t][r][lane];
                        out[(size_t)(row0 + hi8 + r) * outPitch + n] = v / denom[r];
                    }
                }
            }
        }
    }
}

// ---------------- host orchestration ----------------
extern "C" void kernel_launch(void* const* d_in, const int* in_sizes, int n_in,
                              void* d_out, int out_size, void* d_ws, size_t ws_size,
                              hipStream_t stream) {
    (void)in_sizes; (void)n_in; (void)out_size; (void)ws_size;
    const float* x      = (const float*)d_in[0];
    const int*   adj    = (const int*)  d_in[1];
    const float* W1     = (const float*)d_in[2];
    const float* a_src1 = (const float*)d_in[3];
    const float* a_tgt1 = (const float*)d_in[4];
    const float* W2     = (const float*)d_in[5];
    const float* a_src2 = (const float*)d_in[6];
    const float* a_tgt2 = (const float*)d_in[7];

    const int Nn = 6000, Fin = 512, Hid = 64, H1 = 8, C = 40, Cpad = 48;
    const int P = 6016;   // padded node pitch (bf16): tail B-loads in-bounds, 16B aligned

    char* ws = (char*)d_ws;
    size_t off = 0;
    auto alloc = [&](size_t bytes) -> void* {
        void* p = ws + off;
        off = (off + bytes + 255) & ~(size_t)255;
        return p;
    };
    unsigned short* xb    = (unsigned short*)alloc((size_t)Nn * Fin * 2);
    unsigned short* W1Tb  = (unsigned short*)alloc((size_t)H1 * Hid * Fin * 2);
    unsigned short* W2Tb  = (unsigned short*)alloc((size_t)Cpad * Fin * 2);
    unsigned short* Wh1T  = (unsigned short*)alloc((size_t)H1 * Hid * P * 2);
    unsigned short* h1b   = (unsigned short*)alloc((size_t)Nn * Fin * 2);
    unsigned short* Wh2T  = (unsigned short*)alloc((size_t)Cpad * P * 2);
    float* ssrc1 = (float*)alloc((size_t)H1 * Nn * 4);
    float* stgt1 = (float*)alloc((size_t)H1 * Nn * 4);
    float* mmax1 = (float*)alloc((size_t)H1 * Nn * 4);
    float* ssrc2 = (float*)alloc((size_t)Nn * 4);
    float* stgt2 = (float*)alloc((size_t)Nn * 4);
    float* mmax2 = (float*)alloc((size_t)Nn * 4);

    // ---- pack to bf16 ----
    cvt_f32_bf16<<<(unsigned)(((long)Nn * Fin + 255) / 256), 256, 0, stream>>>(x, xb, (long)Nn * Fin);
    pack_W1T<<<(unsigned)(((long)H1 * Fin * Hid + 255) / 256), 256, 0, stream>>>(W1, W1Tb, H1, Fin, Hid);
    pack_W2T<<<(unsigned)(((long)Cpad * Fin + 255) / 256), 256, 0, stream>>>(W2, W2Tb, Fin, C, Cpad);

    // ---- layer 1 ----
    // heads are contiguous 64-row groups of W1T/Wh1T: spread N over blockIdx.y (8*64/16 = 32)
    gemm_bf16_wmma<<<dim3(Nn / 16, (H1 * Hid) / 16), 32, 0, stream>>>(xb, W1Tb, Wh1T, Fin, P);
    sdot_k<<<dim3((Nn + 255) / 256, H1), 256, 0, stream>>>(
        Wh1T, a_src1, a_tgt1, ssrc1, stgt1, Hid, Nn, P, Hid, Hid);
    rowmax_k<8><<<Nn, 256, 0, stream>>>(adj, stgt1, mmax1, Nn);
    gat_attn_wmma<4, 0, 4><<<dim3(Nn / 16, H1), 128, 0, stream>>>(
        adj, ssrc1, stgt1, mmax1, Wh1T, h1b, Nn, P, Fin);

    // ---- layer 2 ----
    gemm_bf16_wmma<<<dim3(Nn / 16, Cpad / 16), 32, 0, stream>>>(h1b, W2Tb, Wh2T, Fin, P);
    sdot_k<<<dim3((Nn + 255) / 256, 1), 256, 0, stream>>>(
        Wh2T, a_src2, a_tgt2, ssrc2, stgt2, C, Nn, P, Cpad, C);
    rowmax_k<1><<<Nn, 256, 0, stream>>>(adj, stgt2, mmax2, Nn);
    gat_attn_wmma<3, 1, 8><<<dim3(Nn / 16, 1), 256, 0, stream>>>(
        adj, ssrc2, stgt2, mmax2, Wh2T, d_out, Nn, P, C);

    // ---- second tuple element: adj passthrough ----
    hipMemcpyAsync((char*)d_out + (size_t)Nn * C * sizeof(float), adj,
                   (size_t)Nn * Nn * sizeof(int), hipMemcpyDeviceToDevice, stream);
}
